// Attention_37958920962105
// MI455X (gfx1250) — matrix operand order repaired and checked by reading, not compiled, not run on hardware
//
#include <hip/hip_runtime.h>
#include <hip/hip_bf16.h>
#include <stdint.h>

// ---------------------------------------------------------------------------
// Types for CDNA5 WMMA (wave32, 16x16x32 bf16 -> f32)
// ---------------------------------------------------------------------------
typedef __attribute__((ext_vector_type(16))) __bf16 v16bf;
typedef __attribute__((ext_vector_type(8)))  __bf16 v8bf;
typedef __attribute__((ext_vector_type(8)))  float  v8f;
typedef int v4i __attribute__((vector_size(16)));            // 16B chunk type
typedef __attribute__((address_space(3))) v4i* lds_v4i_ptr;  // LDS 16B chunk

#define DIMSZ 4096
#define NH    32
#define NKV   8
#define HD    128
#define BB    2
#define SSZ   2048
#define TQ    (BB * SSZ)                 // 4096 tokens
#define FQKV  ((NH + 2 * NKV) * HD)      // 6144 qkv features

#if __has_builtin(__builtin_amdgcn_global_load_async_to_lds_b128)
#define HAVE_ASYNC_LDS 1
#else
#define HAVE_ASYNC_LDS 0
#endif

static __device__ __forceinline__ __bf16 f2bf(float f) { return (__bf16)f; }

static __device__ __forceinline__ v8f wmma_bf16(v16bf a, v16bf b, v8f c) {
    // D = A(16x32 bf16) * B(32x16 bf16) + C(16x16 f32)
    return __builtin_amdgcn_wmma_f32_16x16x32_bf16(
        /*neg_a=*/false, a, /*neg_b=*/false, b,
        /*c_mod=*/(short)0, c, /*reuse_a=*/false, /*reuse_b=*/false);
}

static __device__ __forceinline__ void wait_async0() {
#if __has_builtin(__builtin_amdgcn_s_wait_asynccnt)
    __builtin_amdgcn_s_wait_asynccnt(0);
#else
    asm volatile("s_wait_asynccnt 0" ::: "memory");
#endif
}

// ===========================================================================
// Kernel 1: QKV projection  C[t,f] = sum_d X[t,d] * Wqkv[f,d]
// Tile: BM=128 BN=128 BK=32, 8 waves (2x4). Register double-buffered tile
// loads (next tile's global loads fly during current tile's WMMAs).
// BN==HD so each block maps to exactly one head -> epilogue scatters to
// q / k / v-transposed bf16 buffers.
// ===========================================================================
__global__ void __launch_bounds__(256)
qkv_gemm_kernel(const float* __restrict__ X,       // [TQ, DIMSZ]
                const float* __restrict__ W,       // [FQKV, DIMSZ]
                __bf16* __restrict__ qout,         // [BB, NH, SSZ, HD]
                __bf16* __restrict__ kout,         // [BB, NKV, SSZ, HD]
                __bf16* __restrict__ vtout)        // [BB, NKV, HD, SSZ]
{
    constexpr int BK = 32;
    constexpr int LDA = 40;   // padded bf16 stride (16B-aligned sublanes)
    constexpr int LDB = 48;   // padded bf16 stride (32B-aligned sublanes)
    __shared__ __attribute__((aligned(32))) __bf16 As[128 * LDA];
    __shared__ __attribute__((aligned(32))) __bf16 Bs[128 * LDB];

    const int tid  = threadIdx.x;
    const int lane = tid & 31;
    const int wave = tid >> 5;
    const int wm   = wave >> 2;          // 0..1
    const int wn   = wave & 3;           // 0..3
    const int half = lane >> 4;
    const int n16  = lane & 15;

    const int rowBase = blockIdx.y * 128;
    const int colBase = blockIdx.x * 128;

    v8f acc[4][2] = {};

    // loader mapping: 8 threads/row * 4 f32 each, 32 rows/pass, 4 passes
    const int lrow = tid >> 3;
    const int lcol = (tid & 7) * 4;

    // register-stage tile k0 = 0 (prologue)
    float4 ra[4], rb[4];
#pragma unroll
    for (int p = 0; p < 4; ++p) {
        ra[p] = *(const float4*)(X + (size_t)(rowBase + lrow + p * 32) * DIMSZ + lcol);
        rb[p] = *(const float4*)(W + (size_t)(colBase + lrow + p * 32) * DIMSZ + lcol);
    }

    for (int k0 = 0; k0 < DIMSZ; k0 += BK) {
        // drain staged registers into LDS (converted to bf16)
#pragma unroll
        for (int p = 0; p < 4; ++p) {
            const int r = lrow + p * 32;
            __bf16* da = &As[r * LDA + lcol];
            da[0] = f2bf(ra[p].x); da[1] = f2bf(ra[p].y);
            da[2] = f2bf(ra[p].z); da[3] = f2bf(ra[p].w);
            __bf16* db = &Bs[r * LDB + lcol];
            db[0] = f2bf(rb[p].x); db[1] = f2bf(rb[p].y);
            db[2] = f2bf(rb[p].z); db[3] = f2bf(rb[p].w);
        }
        __syncthreads();

        // issue next tile's global loads; they overlap the WMMAs below
        if (k0 + BK < DIMSZ) {
#pragma unroll
            for (int p = 0; p < 4; ++p) {
                ra[p] = *(const float4*)(X + (size_t)(rowBase + lrow + p * 32) * DIMSZ + (k0 + BK) + lcol);
                rb[p] = *(const float4*)(W + (size_t)(colBase + lrow + p * 32) * DIMSZ + (k0 + BK) + lcol);
            }
        }

        v16bf bfrag[2];
#pragma unroll
        for (int nt = 0; nt < 2; ++nt) {
            const int n = wn * 32 + nt * 16 + n16;
            bfrag[nt] = *(const v16bf*)&Bs[n * LDB + half * 16];
        }
#pragma unroll
        for (int mt = 0; mt < 4; ++mt) {
            const int m = wm * 64 + mt * 16 + n16;
            union { v16bf v; v8bf h[2]; } af;
            af.h[0] = *(const v8bf*)&As[m * LDA + half * 16];
            af.h[1] = *(const v8bf*)&As[m * LDA + half * 16 + 8];
#pragma unroll
            for (int nt = 0; nt < 2; ++nt)
                acc[mt][nt] = wmma_bf16(af.v, bfrag[nt], acc[mt][nt]);
        }
        __syncthreads();
    }

    // Epilogue scatter (block is uniform over one q/k/v head)
#pragma unroll
    for (int mt = 0; mt < 4; ++mt) {
#pragma unroll
        for (int nt = 0; nt < 2; ++nt) {
#pragma unroll
            for (int r = 0; r < 8; ++r) {
                const int t = rowBase + wm * 64 + mt * 16 + r + 8 * half;
                const int f = colBase + wn * 32 + nt * 16 + n16;
                const int b = t >> 11;
                const int s = t & 2047;
                const int d = f & 127;
                const float val = acc[mt][nt][r];
                if (f < DIMSZ) {
                    const int h = f >> 7;
                    qout[(((size_t)(b * NH + h) * SSZ + s) * HD) + d] = f2bf(val);
                } else if (f < DIMSZ + NKV * HD) {
                    const int h = (f - DIMSZ) >> 7;
                    kout[(((size_t)(b * NKV + h) * SSZ + s) * HD) + d] = f2bf(val);
                } else {
                    const int h = (f - DIMSZ - NKV * HD) >> 7;
                    vtout[(((size_t)(b * NKV + h) * HD + d) * SSZ) + s] = f2bf(val);
                }
            }
        }
    }
}

// ===========================================================================
// Kernel 2: RoPE applied in place to bf16 q and k
// ===========================================================================
__global__ void __launch_bounds__(256)
rope_kernel(const float* __restrict__ freqs,   // [SSZ, HD/2, 2]
            __bf16* __restrict__ q,            // [BB, NH, SSZ, HD]
            __bf16* __restrict__ k)            // [BB, NKV, SSZ, HD]
{
    const int t = blockIdx.x;
    const int b = t >> 11;
    const int s = t & 2047;
    for (int idx = threadIdx.x; idx < (NH + NKV) * (HD / 2); idx += 256) {
        const int head = idx >> 6;
        const int j    = idx & 63;
        __bf16* base;
        if (head < NH) base = q + ((size_t)(b * NH + head) * SSZ + s) * HD;
        else           base = k + ((size_t)(b * NKV + (head - NH)) * SSZ + s) * HD;
        const float x0 = (float)base[2 * j];
        const float x1 = (float)base[2 * j + 1];
        const float f0 = freqs[(s * 64 + j) * 2 + 0];
        const float f1 = freqs[(s * 64 + j) * 2 + 1];
        base[2 * j]     = f2bf(x0 * f0 - x1 * f1);
        base[2 * j + 1] = f2bf(x0 * f1 + x1 * f0);
    }
}

// ===========================================================================
// Kernel 3: causal GQA flash attention.
// Grid (S/128, NH, BB), 8 waves; wave owns 16 query rows. Per 32-key block:
// 8 QK^T WMMAs, online softmax (half-wave shfl reductions), P via per-wave
// LDS re-layout (s_wait_dscnt), 8 PV WMMAs against transposed V.
// ===========================================================================
__global__ void __launch_bounds__(256)
attn_kernel(const __bf16* __restrict__ q,    // [BB, NH, SSZ, HD]
            const __bf16* __restrict__ k,    // [BB, NKV, SSZ, HD]
            const __bf16* __restrict__ vt,   // [BB, NKV, HD, SSZ]
            __bf16* __restrict__ y)          // [TQ, DIMSZ] = [b,s,h*HD+d]
{
    constexpr int LDP = 40;
    __shared__ __attribute__((aligned(32))) __bf16 Ps[8 * 16 * LDP];

    const int lane = threadIdx.x & 31;
    const int wave = threadIdx.x >> 5;
    const int half = lane >> 4;
    const int n16  = lane & 15;

    const int h     = blockIdx.y;
    const int b     = blockIdx.z;
    const int qbase = blockIdx.x * 128 + wave * 16;
    const int kvh   = h >> 2;                      // GQA: 32 q heads / 8 kv

    const __bf16* qrow  = q + (((size_t)(b * NH + h) * SSZ) + qbase + n16) * HD;
    const __bf16* kbase = k + ((size_t)(b * NKV + kvh) * SSZ) * HD;
    const __bf16* vbase = vt + ((size_t)(b * NKV + kvh) * HD) * SSZ;
    __bf16* Pl = &Ps[wave * 16 * LDP];

    // Q A-fragments (4 x K=32 covering HD=128)
    v16bf qf[4];
#pragma unroll
    for (int j = 0; j < 4; ++j) {
        union { v16bf v; v8bf h2[2]; } u;
        u.h2[0] = *(const v8bf*)(qrow + j * 32 + half * 16);
        u.h2[1] = *(const v8bf*)(qrow + j * 32 + half * 16 + 8);
        qf[j] = u.v;
    }

    v8f   o[8] = {};
    float mrow[8], lrow[8];
#pragma unroll
    for (int r = 0; r < 8; ++r) { mrow[r] = -3.0e38f; lrow[r] = 0.0f; }

    const float kScale = 0.08838834764831845f * 1.44269504088896341f; // rsqrt(128)*log2(e)

    const int kend = qbase + 16;                   // exclusive causal key limit
    for (int key0 = 0; key0 < kend; key0 += 32) {
        // prefetch next key/value block into cache while this one computes
        if (key0 + 32 < kend) {
            __builtin_prefetch(kbase + (size_t)(key0 + 32 + n16) * HD + half * 64, 0, 1);
            __builtin_prefetch(vbase + (size_t)(wave * 16 + n16) * SSZ + key0 + 32, 0, 1);
        }

        v8f s0 = {}, s1 = {};
#pragma unroll
        for (int j = 0; j < 4; ++j) {
            const v16bf kf0 = *(const v16bf*)(kbase + (size_t)(key0 + n16) * HD + j * 32 + half * 16);
            const v16bf kf1 = *(const v16bf*)(kbase + (size_t)(key0 + 16 + n16) * HD + j * 32 + half * 16);
            s0 = wmma_bf16(qf[j], kf0, s0);
            s1 = wmma_bf16(qf[j], kf1, s1);
        }

        const bool need_mask = (key0 + 31) > qbase;
#pragma unroll
        for (int r = 0; r < 8; ++r) {
            const int qi = qbase + r + 8 * half;
            float a = s0[r] * kScale;
            float c = s1[r] * kScale;
            if (need_mask) {
                if (key0 + n16 > qi)      a = -3.0e38f;
                if (key0 + 16 + n16 > qi) c = -3.0e38f;
            }
            float mx = fmaxf(a, c);
#pragma unroll
            for (int msk = 8; msk >= 1; msk >>= 1)
                mx = fmaxf(mx, __shfl_xor(mx, msk, 32));
            const float mnew  = fmaxf(mrow[r], mx);
            const float alpha = exp2f(mrow[r] - mnew);
            const float e0 = exp2f(a - mnew);
            const float e1 = exp2f(c - mnew);
            float rs = e0 + e1;
#pragma unroll
            for (int msk = 8; msk >= 1; msk >>= 1)
                rs += __shfl_xor(rs, msk, 32);
            lrow[r] = lrow[r] * alpha + rs;
            mrow[r] = mnew;
#pragma unroll
            for (int dt = 0; dt < 8; ++dt)
                o[dt][r] = o[dt][r] * alpha;
            // P in C-layout -> per-wave LDS scratch (bf16)
            Pl[(r + 8 * half) * LDP + n16]      = f2bf(e0);
            Pl[(r + 8 * half) * LDP + 16 + n16] = f2bf(e1);
        }
        asm volatile("s_wait_dscnt 0" ::: "memory");

        // reload P as an A-fragment (16 x 32)
        union { v16bf v; v8bf h2[2]; } pf;
        pf.h2[0] = *(const v8bf*)&Pl[n16 * LDP + half * 16];
        pf.h2[1] = *(const v8bf*)&Pl[n16 * LDP + half * 16 + 8];

#pragma unroll
        for (int dt = 0; dt < 8; ++dt) {
            const v16bf vf = *(const v16bf*)(vbase + (size_t)(dt * 16 + n16) * SSZ + key0 + half * 16);
            o[dt] = wmma_bf16(pf.v, vf, o[dt]);
        }
        asm volatile("s_wait_dscnt 0" ::: "memory");
    }

    // epilogue: O / l -> y (bf16)
#pragma unroll
    for (int r = 0; r < 8; ++r) {
        const float inv = 1.0f / lrow[r];
        const int s = qbase + r + 8 * half;
        __bf16* yrow = y + (size_t)(b * SSZ + s) * DIMSZ + h * HD;
#pragma unroll
        for (int dt = 0; dt < 8; ++dt)
            yrow[dt * 16 + n16] = f2bf(o[dt][r] * inv);
    }
}

// ===========================================================================
// Kernel 4: output projection  Out[t,o] = sum_d Y[t,d] * Wo[o,d]  (f32 out)
// A operand (Y) is bf16 in global memory: moved with ASYNCcnt-tracked
// GLOBAL_LOAD_ASYNC_TO_LDS_B128 into a double-buffered LDS tile when the
// toolchain exposes the builtin; B operand is register-pipelined + converted.
// ===========================================================================
__global__ void __launch_bounds__(256)
out_gemm_kernel(const __bf16* __restrict__ Y,   // [TQ, DIMSZ] bf16
                const float* __restrict__ Wo,   // [DIMSZ, DIMSZ] f32
                float* __restrict__ Out)        // [TQ, DIMSZ] f32
{
    constexpr int BK = 32;
    constexpr int LDA = 40;
    constexpr int LDB = 48;
#if HAVE_ASYNC_LDS
    __shared__ __attribute__((aligned(32))) __bf16 As[2][128 * LDA];
#else
    __shared__ __attribute__((aligned(32))) __bf16 As[1][128 * LDA];
#endif
    __shared__ __attribute__((aligned(32))) __bf16 Bs[128 * LDB];

    const int tid  = threadIdx.x;
    const int lane = tid & 31;
    const int wave = tid >> 5;
    const int wm   = wave >> 2;
    const int wn   = wave & 3;
    const int half = lane >> 4;
    const int n16  = lane & 15;

    const int rowBase = blockIdx.y * 128;
    const int colBase = blockIdx.x * 128;

    v8f acc[4][2] = {};

    const int larow = tid >> 2;          // bf16 A: 4 threads/row * 8 elems (16B)
    const int lacol = (tid & 3) * 8;
    const int lbrow = tid >> 3;          // f32 B: 8 threads/row * 4 elems
    const int lbcol = (tid & 7) * 4;

    // ---- prologue: stage tile k0 = 0 ----
#if HAVE_ASYNC_LDS
#pragma unroll
    for (int p = 0; p < 2; ++p) {
        const int r = larow + p * 64;
        __builtin_amdgcn_global_load_async_to_lds_b128(
            (v4i*)(Y + (size_t)(rowBase + r) * DIMSZ + lacol),
            (lds_v4i_ptr)&As[0][r * LDA + lacol],
            0, 0);
    }
#else
    v8bf ya[2];
#pragma unroll
    for (int p = 0; p < 2; ++p)
        ya[p] = *(const v8bf*)(Y + (size_t)(rowBase + larow + p * 64) * DIMSZ + lacol);
#endif
    float4 rb[4];
#pragma unroll
    for (int p = 0; p < 4; ++p)
        rb[p] = *(const float4*)(Wo + (size_t)(colBase + lbrow + p * 32) * DIMSZ + lbcol);

    int buf = 0;
    for (int k0 = 0; k0 < DIMSZ; k0 += BK) {
        // drain staged B registers into LDS (converted to bf16)
#pragma unroll
        for (int p = 0; p < 4; ++p) {
            const int r = lbrow + p * 32;
            __bf16* db = &Bs[r * LDB + lbcol];
            db[0] = f2bf(rb[p].x); db[1] = f2bf(rb[p].y);
            db[2] = f2bf(rb[p].z); db[3] = f2bf(rb[p].w);
        }
#if HAVE_ASYNC_LDS
        wait_async0();             // this wave's async A tile is in LDS
#else
#pragma unroll
        for (int p = 0; p < 2; ++p)
            *(v8bf*)&As[0][(larow + p * 64) * LDA + lacol] = ya[p];
#endif
        __syncthreads();

        // stage next tile while WMMAs run
        if (k0 + BK < DIMSZ) {
#if HAVE_ASYNC_LDS
#pragma unroll
            for (int p = 0; p < 2; ++p) {
                const int r = larow + p * 64;
                __builtin_amdgcn_global_load_async_to_lds_b128(
                    (v4i*)(Y + (size_t)(rowBase + r) * DIMSZ + (k0 + BK) + lacol),
                    (lds_v4i_ptr)&As[buf ^ 1][r * LDA + lacol],
                    0, 0);
            }
#else
#pragma unroll
            for (int p = 0; p < 2; ++p)
                ya[p] = *(const v8bf*)(Y + (size_t)(rowBase + larow + p * 64) * DIMSZ + (k0 + BK) + lacol);
#endif
#pragma unroll
            for (int p = 0; p < 4; ++p)
                rb[p] = *(const float4*)(Wo + (size_t)(colBase + lbrow + p * 32) * DIMSZ + (k0 + BK) + lbcol);
        }

        v16bf bfrag[2];
#pragma unroll
        for (int nt = 0; nt < 2; ++nt) {
            const int n = wn * 32 + nt * 16 + n16;
            bfrag[nt] = *(const v16bf*)&Bs[n * LDB + half * 16];
        }
#pragma unroll
        for (int mt = 0; mt < 4; ++mt) {
            const int m = wm * 64 + mt * 16 + n16;
            union { v16bf v; v8bf h[2]; } af;
            af.h[0] = *(const v8bf*)&As[buf][m * LDA + half * 16];
            af.h[1] = *(const v8bf*)&As[buf][m * LDA + half * 16 + 8];
#pragma unroll
            for (int nt = 0; nt < 2; ++nt)
                acc[mt][nt] = wmma_bf16(af.v, bfrag[nt], acc[mt][nt]);
        }
        __syncthreads();
#if HAVE_ASYNC_LDS
        buf ^= 1;
#endif
    }

#pragma unroll
    for (int mt = 0; mt < 4; ++mt) {
#pragma unroll
        for (int nt = 0; nt < 2; ++nt) {
#pragma unroll
            for (int r = 0; r < 8; ++r) {
                const int t = rowBase + wm * 64 + mt * 16 + r + 8 * half;
                const int oc = colBase + wn * 32 + nt * 16 + n16;
                Out[(size_t)t * DIMSZ + oc] = acc[mt][nt][r];
            }
        }
    }
}

// ===========================================================================
// Host launcher
// ===========================================================================
extern "C" void kernel_launch(void* const* d_in, const int* in_sizes, int n_in,
                              void* d_out, int out_size, void* d_ws, size_t ws_size,
                              hipStream_t stream) {
    (void)in_sizes; (void)n_in; (void)out_size; (void)ws_size;

    const float* x     = (const float*)d_in[0];   // [BB, SSZ, DIMSZ]
    const float* freqs = (const float*)d_in[1];   // [SSZ, HD/2, 2]
    // d_in[2] mask_cache, d_in[3] input_pos: implicit in causal kernel
    const float* wqkv  = (const float*)d_in[4];   // [FQKV, DIMSZ]
    const float* wo    = (const float*)d_in[5];   // [DIMSZ, DIMSZ]

    char* ws = (char*)d_ws;
    __bf16* qb  = (__bf16*)(ws);                          // 32 MB
    __bf16* kb  = (__bf16*)(ws + (size_t)33554432);       //  8 MB
    __bf16* vtb = (__bf16*)(ws + (size_t)41943040);       //  8 MB
    __bf16* yb  = (__bf16*)(ws + (size_t)50331648);       // 32 MB

    dim3 g1(FQKV / 128, TQ / 128);       // 48 x 32
    qkv_gemm_kernel<<<g1, 256, 0, stream>>>(x, wqkv, qb, kb, vtb);

    rope_kernel<<<TQ, 256, 0, stream>>>(freqs, qb, kb);

    dim3 g3(SSZ / 128, NH, BB);          // 16 x 32 x 2
    attn_kernel<<<g3, 256, 0, stream>>>(qb, kb, vtb, yb);

    dim3 g4(DIMSZ / 128, TQ / 128);      // 32 x 32
    out_gemm_kernel<<<g4, 256, 0, stream>>>(yb, wo, (float*)d_out);
}
